// CausalAttention_27419071217734
// MI455X (gfx1250) — matrix operand order repaired and checked by reading, not compiled, not run on hardware
//
#include <hip/hip_runtime.h>

typedef __attribute__((ext_vector_type(16))) _Float16 v16h;
typedef __attribute__((ext_vector_type(8)))  _Float16 v8h;
typedef __attribute__((ext_vector_type(8)))  float    v8f;
typedef __attribute__((ext_vector_type(4)))  int      v4i;

#define AS1 __attribute__((address_space(1)))
#define AS3 __attribute__((address_space(3)))

#define HD  256   // hidden
#define SEQ 4096  // sequence
#define NB  4     // batch

#define HAS_ASYNC_LDS __has_builtin(__builtin_amdgcn_global_load_async_to_lds_b128)

__device__ __forceinline__ v8f wmma16(v16h a, v16h b, v8f c) {
  // D = A(16x32 f16) * B(32x16 f16) + C(16x16 f32)
  return __builtin_amdgcn_wmma_f32_16x16x32_f16(false, a, false, b, (short)0, c, false, false);
}

__device__ __forceinline__ void async_cp16(_Float16* lds, const _Float16* g) {
#if HAS_ASYNC_LDS
  __builtin_amdgcn_global_load_async_to_lds_b128(
      (AS1 v4i*)(AS1 void*)(void*)g,
      (AS3 v4i*)(AS3 void*)(void*)lds, 0, 0);
#endif
}

__device__ __forceinline__ void wait_async_done() {
#if HAS_ASYNC_LDS
#if __has_builtin(__builtin_amdgcn_s_wait_asynccnt)
  __builtin_amdgcn_s_wait_asynccnt(0);
#else
  asm volatile("s_wait_asynccnt 0x0" ::: "memory");
#endif
#endif
}

// ---------------------------------------------------------------------------
// Kernel 1: convert + transpose weights: W[in][out] f32 -> Wt[mat][out][in] f16
// ---------------------------------------------------------------------------
__global__ __launch_bounds__(256)
void prep_weights(const float* __restrict__ Wq, const float* __restrict__ Wk,
                  const float* __restrict__ Wv, _Float16* __restrict__ Wt) {
  int idx = blockIdx.x * 256 + threadIdx.x;        // [0, 3*256*256)
  int mat = idx >> 16;
  int rem = idx & 65535;
  int in  = rem >> 8;
  int out = rem & 255;
  const float* W = (mat == 0) ? Wq : ((mat == 1) ? Wk : Wv);
  Wt[(size_t)mat * 65536 + (size_t)out * 256 + in] = (_Float16)W[in * 256 + out];
}

// ---------------------------------------------------------------------------
// Kernel 2: fused QKV projection. Each wave: 16 rows x 256 cols x 3 matrices.
// Q,K stored row-major f16 [B*S][H]; V stored transposed f16 [B][H][S].
// ---------------------------------------------------------------------------
__global__ __launch_bounds__(128)
void proj_kernel(const float* __restrict__ x, const _Float16* __restrict__ Wt,
                 const float* __restrict__ bq, const float* __restrict__ bk,
                 const float* __restrict__ bv,
                 _Float16* __restrict__ Q, _Float16* __restrict__ K,
                 _Float16* __restrict__ Vt) {
  const int lane = threadIdx.x & 31;
  const int wave = threadIdx.x >> 5;
  const int m0   = blockIdx.x * 64 + wave * 16;    // global row in [0, B*S)
  const int lq   = lane & 15;
  const int lh   = lane >> 4;

  v16h xf[8];
  const float* xrow = x + (size_t)(m0 + lq) * HD;
  #pragma unroll
  for (int hc = 0; hc < 8; ++hc) {
    const int kb = hc * 32 + lh * 16;
    const float4* p0 = (const float4*)(xrow + kb);
    const float4* p1 = (const float4*)(xrow + kb + 8);
    float4 f0 = p0[0], f1 = p0[1], f2 = p1[0], f3 = p1[1];
    xf[hc][0]  = (_Float16)f0.x; xf[hc][1]  = (_Float16)f0.y;
    xf[hc][2]  = (_Float16)f0.z; xf[hc][3]  = (_Float16)f0.w;
    xf[hc][4]  = (_Float16)f1.x; xf[hc][5]  = (_Float16)f1.y;
    xf[hc][6]  = (_Float16)f1.z; xf[hc][7]  = (_Float16)f1.w;
    xf[hc][8]  = (_Float16)f2.x; xf[hc][9]  = (_Float16)f2.y;
    xf[hc][10] = (_Float16)f2.z; xf[hc][11] = (_Float16)f2.w;
    xf[hc][12] = (_Float16)f3.x; xf[hc][13] = (_Float16)f3.y;
    xf[hc][14] = (_Float16)f3.z; xf[hc][15] = (_Float16)f3.w;
  }

  for (int mat = 0; mat < 3; ++mat) {
    const _Float16* Wm  = Wt + (size_t)mat * 65536;
    const float*    bia = (mat == 0) ? bq : ((mat == 1) ? bk : bv);
    for (int nt = 0; nt < 16; ++nt) {
      const int ncol = nt * 16 + lq;
      const float bvv = bia[ncol];
      v8f acc;
      #pragma unroll
      for (int j = 0; j < 8; ++j) acc[j] = bvv;
      #pragma unroll
      for (int hc = 0; hc < 8; ++hc) {
        v16h wf = *(const v16h*)(Wm + (size_t)ncol * 256 + hc * 32 + lh * 16);
        acc = wmma16(xf[hc], wf, acc);
      }
      if (mat < 2) {
        _Float16* Out = (mat == 0) ? Q : K;
        #pragma unroll
        for (int j = 0; j < 8; ++j) {
          const int row = m0 + j + lh * 8;
          Out[(size_t)row * HD + ncol] = (_Float16)acc[j];
        }
      } else {
        #pragma unroll
        for (int j = 0; j < 8; ++j) {
          const int row = m0 + j + lh * 8;
          const int b = row >> 12, s = row & (SEQ - 1);
          Vt[(size_t)b * HD * SEQ + (size_t)ncol * SEQ + s] = (_Float16)acc[j];
        }
      }
    }
  }
}

// ---------------------------------------------------------------------------
// Kernel 3: causal flash attention, double-buffered LDS K/V tiles with
// async global->LDS staging (ASYNCcnt) when available.
// Block = 4 waves = 64 queries; waves share each 32-key K/V chunk via LDS.
// ---------------------------------------------------------------------------
#define KSTR 264   // K LDS row pitch in halves (32 rows)
#define VSTR 40    // V LDS row pitch in halves (256 rows)
#define KBUF (32 * KSTR)
#define VBUF (256 * VSTR)

// stage one 32-key chunk (K: 32x256, Vt: 256x32) into LDS buffer
__device__ __forceinline__ void stage_chunk(int k0, int tid,
                                            const _Float16* __restrict__ Kb,
                                            const _Float16* __restrict__ Vb,
                                            _Float16* Kl, _Float16* Vl) {
  const int kr = tid >> 2;           // K row 0..31
  const int kc = (tid & 3) * 64;     // 64-half slice
  const int vr = tid * 2;            // V rows 2t, 2t+1
  const _Float16* gk = Kb + (size_t)(k0 + kr) * HD + kc;
  _Float16*       sk = Kl + kr * KSTR + kc;
  const _Float16* gv = Vb + (size_t)vr * SEQ + k0;
  _Float16*       sv = Vl + vr * VSTR;
#if HAS_ASYNC_LDS
  #pragma unroll
  for (int i = 0; i < 8; ++i) async_cp16(sk + i * 8, gk + i * 8);
  #pragma unroll
  for (int i = 0; i < 4; ++i) async_cp16(sv + i * 8, gv + i * 8);
  #pragma unroll
  for (int i = 0; i < 4; ++i) async_cp16(sv + VSTR + i * 8, gv + SEQ + i * 8);
#else
  v16h t0 = *(const v16h*)(gk);
  v16h t1 = *(const v16h*)(gk + 16);
  v16h t2 = *(const v16h*)(gk + 32);
  v16h t3 = *(const v16h*)(gk + 48);
  *(v8h*)(sk)      = t0.lo; *(v8h*)(sk + 8)  = t0.hi;
  *(v8h*)(sk + 16) = t1.lo; *(v8h*)(sk + 24) = t1.hi;
  *(v8h*)(sk + 32) = t2.lo; *(v8h*)(sk + 40) = t2.hi;
  *(v8h*)(sk + 48) = t3.lo; *(v8h*)(sk + 56) = t3.hi;
  v16h u0 = *(const v16h*)(gv);
  v16h u1 = *(const v16h*)(gv + 16);
  v16h u2 = *(const v16h*)(gv + SEQ);
  v16h u3 = *(const v16h*)(gv + SEQ + 16);
  *(v8h*)(sv)             = u0.lo; *(v8h*)(sv + 8)         = u0.hi;
  *(v8h*)(sv + 16)        = u1.lo; *(v8h*)(sv + 24)        = u1.hi;
  *(v8h*)(sv + VSTR)      = u2.lo; *(v8h*)(sv + VSTR + 8)  = u2.hi;
  *(v8h*)(sv + VSTR + 16) = u3.lo; *(v8h*)(sv + VSTR + 24) = u3.hi;
#endif
}

__global__ __launch_bounds__(128)
void attn_kernel(const _Float16* __restrict__ Q, const _Float16* __restrict__ Kp,
                 const _Float16* __restrict__ Vt, float* __restrict__ out) {
  __shared__ _Float16 Klds[2 * KBUF];   // 2 x 16.5 KB
  __shared__ _Float16 Vlds[2 * VBUF];   // 2 x 20 KB

  const int tid  = threadIdx.x;
  const int lane = tid & 31;
  const int wave = tid >> 5;
  const int b    = blockIdx.x >> 6;             // 64 blocks per batch
  const int Q0   = (blockIdx.x & 63) * 64;      // block query base
  const int q0   = Q0 + wave * 16;              // wave query base
  const int lq   = lane & 15;
  const int lh   = lane >> 4;

  const _Float16* Qb = Q  + (size_t)b * SEQ * HD;
  const _Float16* Kb = Kp + (size_t)b * SEQ * HD;
  const _Float16* Vb = Vt + (size_t)b * HD * SEQ;

  // Q as B-fragments: lane holds query col q0+lq
  v16h qf[8];
  const _Float16* qrow = Qb + (size_t)(q0 + lq) * HD;
  #pragma unroll
  for (int hc = 0; hc < 8; ++hc)
    qf[hc] = *(const v16h*)(qrow + hc * 32 + lh * 16);

  v8f acc[16];
  #pragma unroll
  for (int t = 0; t < 16; ++t)
    #pragma unroll
    for (int j = 0; j < 8; ++j) acc[t][j] = 0.0f;

  float m = -1e30f, l = 0.0f;
  const int q = q0 + lq;
  const int kend = Q0 + 63;

  // prologue: stage chunk 0 into buffer 0
  stage_chunk(0, tid, Kb, Vb, Klds, Vlds);

  int buf = 0;
  for (int k0 = 0; k0 <= kend; k0 += 32) {
    wait_async_done();     // this wave's staging of current buffer complete
    __syncthreads();       // everyone's staging visible; prior buffer reads done

    if (k0 + 32 <= kend)   // prefetch next chunk into the other buffer
      stage_chunk(k0 + 32, tid, Kb, Vb, Klds + (buf ^ 1) * KBUF, Vlds + (buf ^ 1) * VBUF);

    if (k0 <= q0 + 15) {   // wave-uniform: skip chunks past this wave's diagonal
      const _Float16* Kl = Klds + buf * KBUF;
      const _Float16* Vl = Vlds + buf * VBUF;

      // ---- S^T tiles from LDS: rows = keys, cols = queries ----
      v8f s0, s1;
      #pragma unroll
      for (int j = 0; j < 8; ++j) { s0[j] = 0.0f; s1[j] = 0.0f; }
      #pragma unroll
      for (int hc = 0; hc < 8; ++hc) {
        const int kb = hc * 32 + lh * 16;
        const _Float16* r0 = Kl + lq * KSTR + kb;
        v8h a0 = *(const v8h*)r0;
        v8h a1 = *(const v8h*)(r0 + 8);
        v16h kf0;
        #pragma unroll
        for (int i = 0; i < 8; ++i) { kf0[i] = a0[i]; kf0[8 + i] = a1[i]; }
        s0 = wmma16(kf0, qf[hc], s0);

        const _Float16* r1 = Kl + (16 + lq) * KSTR + kb;
        v8h c0 = *(const v8h*)r1;
        v8h c1 = *(const v8h*)(r1 + 8);
        v16h kf1;
        #pragma unroll
        for (int i = 0; i < 8; ++i) { kf1[i] = c0[i]; kf1[8 + i] = c1[i]; }
        s1 = wmma16(kf1, qf[hc], s1);
      }

      // ---- causal mask near the diagonal (wave-uniform branch) ----
      if (k0 + 31 > q0) {
        #pragma unroll
        for (int j = 0; j < 8; ++j) {
          const int key0 = k0 + lh * 8 + j;
          if (key0 > q)      s0[j] = -3.0e38f;
          if (key0 + 16 > q) s1[j] = -3.0e38f;
        }
      }

      // ---- online softmax (lane = query layout) ----
      float mc = s0[0];
      #pragma unroll
      for (int j = 1; j < 8; ++j) mc = fmaxf(mc, s0[j]);
      #pragma unroll
      for (int j = 0; j < 8; ++j) mc = fmaxf(mc, s1[j]);
      mc = fmaxf(mc, __shfl_xor(mc, 16));
      const float mnew  = fmaxf(m, mc);
      const float scale = __expf(m - mnew);

      v16h pf;     // P fragment: halves 0..7 <- tile0, 8..15 <- tile1
      float ls = 0.0f;
      #pragma unroll
      for (int j = 0; j < 8; ++j) {
        float p = __expf(s0[j] - mnew); ls += p; pf[j]     = (_Float16)p;
        p       = __expf(s1[j] - mnew); ls += p; pf[8 + j] = (_Float16)p;
      }
      ls += __shfl_xor(ls, 16);
      l = l * scale + ls;
      m = mnew;

      float sc[8];
      #pragma unroll
      for (int j = 0; j < 8; ++j) sc[j] = __shfl(scale, j + lh * 8);
      #pragma unroll
      for (int t = 0; t < 16; ++t)
        #pragma unroll
        for (int j = 0; j < 8; ++j) acc[t][j] *= sc[j];

      // ---- O += P * V, B-frags from LDS ----
      #pragma unroll
      for (int t = 0; t < 16; ++t) {
        const _Float16* vp = Vl + (t * 16 + lq) * VSTR + lh * 8;
        v8h p0 = *(const v8h*)vp;
        v8h p1 = *(const v8h*)(vp + 16);
        v16h vf;
        #pragma unroll
        for (int i = 0; i < 8; ++i) { vf[i] = p0[i]; vf[8 + i] = p1[i]; }
        acc[t] = wmma16(pf, vf, acc[t]);
      }
    }
    buf ^= 1;
  }

  // ---- normalize and store fp32 output ----
  const float rinv = 1.0f / l;
  float rn[8];
  #pragma unroll
  for (int j = 0; j < 8; ++j) rn[j] = __shfl(rinv, j + lh * 8);
  float* ob = out + (size_t)b * SEQ * HD;
  #pragma unroll
  for (int t = 0; t < 16; ++t)
    #pragma unroll
    for (int j = 0; j < 8; ++j)
      ob[(size_t)(q0 + j + lh * 8) * HD + t * 16 + lq] = acc[t][j] * rn[j];
}

// ---------------------------------------------------------------------------
extern "C" void kernel_launch(void* const* d_in, const int* in_sizes, int n_in,
                              void* d_out, int out_size, void* d_ws, size_t ws_size,
                              hipStream_t stream) {
  const float* x  = (const float*)d_in[0];
  const float* Wq = (const float*)d_in[1];
  const float* bq = (const float*)d_in[2];
  const float* Wk = (const float*)d_in[3];
  const float* bk = (const float*)d_in[4];
  const float* Wv = (const float*)d_in[5];
  const float* bv = (const float*)d_in[6];
  float* out = (float*)d_out;

  char* ws = (char*)d_ws;
  _Float16* Wt = (_Float16*)ws;                       // 3*256*256 f16 = 384 KB
  _Float16* Q  = (_Float16*)(ws + 393216);            // 4*4096*256 f16 = 8 MB
  _Float16* K  = Q + (size_t)NB * SEQ * HD;           // 8 MB
  _Float16* Vt = K + (size_t)NB * SEQ * HD;           // 8 MB (transposed V)

  prep_weights<<<768, 256, 0, stream>>>(Wq, Wk, Wv, Wt);
  proj_kernel<<<(NB * SEQ) / 64, 128, 0, stream>>>(x, Wt, bq, bk, bv, Q, K, Vt);
  attn_kernel<<<NB * (SEQ / 64), 128, 0, stream>>>(Q, K, Vt, out);
}
